// ATMixer_34454227649121
// MI455X (gfx1250) — matrix operand order repaired and checked by reading, not compiled, not run on hardware
//
#include <hip/hip_runtime.h>
#include <hip/hip_bf16.h>

// ---------------------------------------------------------------------------
// ATMixer forward for MI455X (gfx1250, wave32, WMMA).
// Heavy GEMMs: bf16 V_WMMA_F32_16X16X32_BF16, f32 accumulate.
// Weight staging uses GLOBAL_LOAD_ASYNC_TO_LDS_B128 (ASYNCcnt) so staging
// needs no VGPRs (no spills) and never interacts with the DScnt waits of the
// B-fragment ds_loads feeding the WMMAs.
// ---------------------------------------------------------------------------

typedef __attribute__((ext_vector_type(16))) __bf16 v16bf;
typedef __attribute__((ext_vector_type(8)))  float  v8f;

#define BB 16
#define NN 4096
#define CC 256
#define HH 64
#define MM (BB * NN)          // 65536 rows
#define SW_STRIDE 72          // padded LDS row stride (elements)

// ---- bf16 <-> f32 helpers (bit-level; buffers carried as unsigned short) ---
__device__ __forceinline__ unsigned short bfbits(float f) {
    union { __hip_bfloat16 h; unsigned short u; } cv;
    cv.h = __float2bfloat16(f);
    return cv.u;
}
__device__ __forceinline__ float bfval(unsigned short u) {
    union { unsigned int u; float f; } r;
    r.u = ((unsigned int)u) << 16;
    return r.f;
}

__device__ __forceinline__ void wait_async0() {
#if __has_builtin(__builtin_amdgcn_s_wait_asynccnt)
    __builtin_amdgcn_s_wait_asynccnt(0);
#else
    asm volatile("s_wait_asynccnt 0x0" ::: "memory");
#endif
}

// global -> LDS async DMA, 16 bytes per lane (ASYNCcnt-tracked, no VGPR data)
__device__ __forceinline__ void async_copy_b128(unsigned lds_off,
                                                const void* gaddr) {
    asm volatile("global_load_async_to_lds_b128 %0, %1, off"
                 :: "v"(lds_off), "v"(gaddr) : "memory");
}

// ---------------------------------------------------------------------------
// Kernel 1: cast the four [C,C] f32 weights to bf16, transposed -> Wt[n][k]
// ---------------------------------------------------------------------------
__global__ __launch_bounds__(256)
void prep_weights(const float* __restrict__ w0, const float* __restrict__ w1,
                  const float* __restrict__ w2, const float* __restrict__ w3,
                  unsigned short* __restrict__ o0, unsigned short* __restrict__ o1,
                  unsigned short* __restrict__ o2, unsigned short* __restrict__ o3) {
    const float* w = (blockIdx.y == 0) ? w0 : (blockIdx.y == 1) ? w1
                   : (blockIdx.y == 2) ? w2 : w3;
    unsigned short* o = (blockIdx.y == 0) ? o0 : (blockIdx.y == 1) ? o1
                      : (blockIdx.y == 2) ? o2 : o3;
    int i = blockIdx.x * 256 + threadIdx.x;   // i = n*256 + k
    int n = i >> 8, k = i & 255;
    o[i] = bfbits(w[k * CC + n]);
}

// ---------------------------------------------------------------------------
// Kernel 2: LayerNorm over C + cast (wave-per-row, wave32 shuffle reduce).
// ---------------------------------------------------------------------------
__global__ __launch_bounds__(256)
void ln_cast(const float* __restrict__ x, const float* __restrict__ g,
             const float* __restrict__ be,
             unsigned short* __restrict__ xn, unsigned short* __restrict__ xb) {
    const int wave = threadIdx.x >> 5, lane = threadIdx.x & 31;
    const int row  = blockIdx.x * 8 + wave;
    const float* xr = x + (size_t)row * CC;
    const int c0 = lane * 4, c1 = 128 + lane * 4;

    float4 v0 = *(const float4*)(xr + c0);
    float4 v1 = *(const float4*)(xr + c1);

    float s = v0.x + v0.y + v0.z + v0.w + v1.x + v1.y + v1.z + v1.w;
#pragma unroll
    for (int m = 16; m; m >>= 1) s += __shfl_xor(s, m, 32);
    const float mu = s * (1.0f / 256.0f);

    float d0x = v0.x - mu, d0y = v0.y - mu, d0z = v0.z - mu, d0w = v0.w - mu;
    float d1x = v1.x - mu, d1y = v1.y - mu, d1z = v1.z - mu, d1w = v1.w - mu;
    float vs = d0x*d0x + d0y*d0y + d0z*d0z + d0w*d0w
             + d1x*d1x + d1y*d1y + d1z*d1z + d1w*d1w;
#pragma unroll
    for (int m = 16; m; m >>= 1) vs += __shfl_xor(vs, m, 32);
    const float rinv = rsqrtf(vs * (1.0f / 256.0f) + 1e-6f);

    float4 g0v = *(const float4*)(g + c0),  g1v = *(const float4*)(g + c1);
    float4 b0v = *(const float4*)(be + c0), b1v = *(const float4*)(be + c1);

    ushort4 pk;
    unsigned short* xnp = xn + (size_t)row * CC;
    unsigned short* xbp = xb + (size_t)row * CC;

    pk.x = bfbits(d0x * rinv * g0v.x + b0v.x);
    pk.y = bfbits(d0y * rinv * g0v.y + b0v.y);
    pk.z = bfbits(d0z * rinv * g0v.z + b0v.z);
    pk.w = bfbits(d0w * rinv * g0v.w + b0v.w);
    *(ushort4*)(xnp + c0) = pk;
    pk.x = bfbits(d1x * rinv * g1v.x + b1v.x);
    pk.y = bfbits(d1y * rinv * g1v.y + b1v.y);
    pk.z = bfbits(d1z * rinv * g1v.z + b1v.z);
    pk.w = bfbits(d1w * rinv * g1v.w + b1v.w);
    *(ushort4*)(xnp + c1) = pk;

    pk.x = bfbits(v0.x); pk.y = bfbits(v0.y); pk.z = bfbits(v0.z); pk.w = bfbits(v0.w);
    *(ushort4*)(xbp + c0) = pk;
    pk.x = bfbits(v1.x); pk.y = bfbits(v1.y); pk.z = bfbits(v1.z); pk.w = bfbits(v1.w);
    *(ushort4*)(xbp + c1) = pk;
}

// ---------------------------------------------------------------------------
// Kernel 3: WMMA GEMM  Out[M,256] = A[M,256](bf16) x W[256,256] (+bias)
// Block: 256 threads = 8 waves over a 256-row x 128-col tile (grid.y = 2).
// Each wave: 2 M-tiles x 8 N-tiles; every B fragment feeds two WMMAs.
// Next-chunk weights stream global->LDS via async DMA (ASYNCcnt), fully
// overlapped with the 32 WMMAs; B frags pipelined from LDS (distance-2).
//
// Fragment layouts per CDNA5 ISA 7.12.2 (wave32):
//   A 16x32 bf16 : lane L -> m=L&15, half=L>>4; elem e -> K=(e&7)+(e>=8?16:0)+8*half
//   B 32x16 bf16 : lane L -> n=L&15, half=L>>4; elem e -> K=(e&7)+(e>=8?16:0)+8*half
//   C/D 16x16 f32: lane L -> n=L&15; vgpr r -> m=r+8*half
// ---------------------------------------------------------------------------
union FragB { v16bf v; int4 i[2]; };

__global__ __launch_bounds__(256)
void gemm_bf16(const unsigned short* __restrict__ A,
               const unsigned short* __restrict__ Wt,
               const float* __restrict__ bias,
               void* __restrict__ Out, int storeBf16) {
    __shared__ __attribute__((aligned(16))) unsigned short sW[2][128 * SW_STRIDE];

    const int tid  = threadIdx.x;
    const int lane = tid & 31, wave = tid >> 5;
    const int half = lane >> 4, l16 = lane & 15;
    const int rowBase = blockIdx.x * 256 + wave * 32;
    const int colBase = blockIdx.y * 128;
    const unsigned short* arow0 = A + (size_t)(rowBase + l16) * CC;
    const unsigned short* arow1 = arow0 + 16 * CC;

    v8f acc[2][8];
    const v8f vzero = {0.f, 0.f, 0.f, 0.f, 0.f, 0.f, 0.f, 0.f};
#pragma unroll
    for (int s = 0; s < 2; s++)
#pragma unroll
        for (int t = 0; t < 8; t++) acc[s][t] = vzero;

    const int rW = tid >> 3;          // 0..31 (staging row group)
    const int cW = (tid & 7) * 8;     // 0,8,..,56 within 64-wide K chunk

    // async global->LDS staging of one 128x64 weight chunk (no VGPR data)
    auto stageAsync = [&](int k0, unsigned short* dst) {
#pragma unroll
        for (int j = 0; j < 4; j++) {
            const unsigned short* src =
                Wt + (size_t)(colBase + j * 32 + rW) * CC + k0 + cW;
            const unsigned lds_off =
                (unsigned)(size_t)(const void*)&dst[(j * 32 + rW) * SW_STRIDE + cW];
            async_copy_b128(lds_off, src);
        }
    };

    stageAsync(0, sW[0]);
    wait_async0();
    __syncthreads();

    for (int c = 0; c < 4; c++) {                 // K chunks of 64
        const int k0 = c * 64;
        const unsigned short* sb = sW[c & 1];

        if (c < 3) stageAsync(k0 + 64, sW[(c + 1) & 1]);  // overlaps compute

        FragB a00, a10, a01, a11;
        a00.i[0] = *(const int4*)(arow0 + k0      + 8 * half);
        a00.i[1] = *(const int4*)(arow0 + k0 + 16 + 8 * half);
        a10.i[0] = *(const int4*)(arow1 + k0      + 8 * half);
        a10.i[1] = *(const int4*)(arow1 + k0 + 16 + 8 * half);
        a01.i[0] = *(const int4*)(arow0 + k0 + 32 + 8 * half);
        a01.i[1] = *(const int4*)(arow0 + k0 + 48 + 8 * half);
        a11.i[0] = *(const int4*)(arow1 + k0 + 32 + 8 * half);
        a11.i[1] = *(const int4*)(arow1 + k0 + 48 + 8 * half);

        auto loadB = [&](FragB& f, int t, int kk) {
            const unsigned short* bp =
                &sb[(t * 16 + l16) * SW_STRIDE + kk + 8 * half];
            f.i[0] = *(const int4*)(bp);
            f.i[1] = *(const int4*)(bp + 16);
        };

#pragma unroll
        for (int kk = 0; kk < 64; kk += 32) {
            const FragB& aLo = (kk == 0) ? a00 : a01;
            const FragB& aHi = (kk == 0) ? a10 : a11;
            FragB bq[3];
            loadB(bq[0], 0, kk);
            loadB(bq[1], 1, kk);
#pragma unroll
            for (int t = 0; t < 8; t++) {
                if (t < 6) loadB(bq[(t + 2) % 3], t + 2, kk);  // distance-2
                const FragB& cur = bq[t % 3];
                acc[0][t] = __builtin_amdgcn_wmma_f32_16x16x32_bf16(
                    false, aLo.v, false, cur.v, (short)0, acc[0][t], false, false);
                acc[1][t] = __builtin_amdgcn_wmma_f32_16x16x32_bf16(
                    false, aHi.v, false, cur.v, (short)0, acc[1][t], false, false);
            }
        }

        if (c < 3) wait_async0();     // staged chunk landed in LDS
        __syncthreads();
    }

#pragma unroll
    for (int s = 0; s < 2; s++) {
#pragma unroll
        for (int t = 0; t < 8; t++) {
            const int col = colBase + t * 16 + l16;
            const float bv = bias ? bias[col] : 0.0f;
#pragma unroll
            for (int r = 0; r < 8; r++) {
                const int row = rowBase + s * 16 + r + 8 * half;
                const float val = acc[s][t][r] + bv;
                if (storeBf16)
                    ((unsigned short*)Out)[(size_t)row * CC + col] = bfbits(val);
                else
                    ((float*)Out)[(size_t)row * CC + col] = val;
            }
        }
    }
}

// ---------------------------------------------------------------------------
// Kernel 4: bilinear gather along N with zero padding + mask, emit bf16.
// ---------------------------------------------------------------------------
__global__ __launch_bounds__(256)
void gather_bilinear(const float* __restrict__ x, const float* __restrict__ off,
                     const int* __restrict__ mask, unsigned short* __restrict__ samp) {
    const int i  = blockIdx.x * 256 + threadIdx.x;   // over B*N*C
    const int ch = i & 255;
    const int n  = (i >> 8) & 4095;
    const int b  = i >> 20;

    const float pos = (float)n + off[i];
    const float p0f = floorf(pos);
    const float w1  = pos - p0f;
    const int i0 = (int)p0f, i1 = i0 + 1;

    const size_t base = (size_t)b * NN * CC + ch;
    const float v0 = (i0 >= 0 && i0 < NN) ? x[base + (size_t)i0 * CC] : 0.0f;
    const float v1 = (i1 >= 0 && i1 < NN) ? x[base + (size_t)i1 * CC] : 0.0f;
    const float m  = (float)mask[b * NN + n];
    samp[i] = bfbits((v0 * (1.0f - w1) + v1 * w1) * m);
}

// ---------------------------------------------------------------------------
// Kernel 5a/5b: masked sum pool, two-level: partials + f32 atomics.
// ---------------------------------------------------------------------------
__global__ __launch_bounds__(256)
void zero_f32(float* __restrict__ a) {
    a[blockIdx.x * 256 + threadIdx.x] = 0.0f;
}

__global__ __launch_bounds__(256)
void pool_partial(const float* __restrict__ s, const unsigned short* __restrict__ cb,
                  const int* __restrict__ mask, float* __restrict__ a) {
    const int b = blockIdx.y, t = threadIdx.x;
    const int n0 = blockIdx.x * 256;
    float acc = 0.0f;
    for (int n = n0; n < n0 + 256; n++) {
        const float m = (float)mask[b * NN + n];
        const size_t idx = (size_t)b * NN * CC + (size_t)n * CC + t;
        acc += (s[idx] + bfval(cb[idx])) * m;
    }
    atomicAdd(&a[b * CC + t], acc);
}

// ---------------------------------------------------------------------------
// Kernel 6: fusion MLP (computes mask-sum itself): a = pooled/msum;
// gelu(a@fc1+b1) @ fc2 + b2 -> pairwise softmax gates.
// ---------------------------------------------------------------------------
__global__ __launch_bounds__(256)
void fusion_mlp(const float* __restrict__ apool, const int* __restrict__ mask,
                const float* __restrict__ fc1_w, const float* __restrict__ fc1_b,
                const float* __restrict__ fc2_w, const float* __restrict__ fc2_b,
                float* __restrict__ g0, float* __restrict__ g1) {
    const int b = blockIdx.x, t = threadIdx.x;
    __shared__ float sa[CC];
    __shared__ float sh[HH];
    __shared__ float sm[CC];

    float lm = 0.0f;
    for (int n = t; n < NN; n += 256) lm += (float)mask[b * NN + n];
    sm[t] = lm;
    __syncthreads();
#pragma unroll
    for (int st = 128; st > 0; st >>= 1) {
        if (t < st) sm[t] += sm[t + st];
        __syncthreads();
    }
    const float msum = sm[0];
    sa[t] = apool[b * CC + t] / msum;
    __syncthreads();

    if (t < HH) {
        float acc = fc1_b[t];
        for (int k = 0; k < CC; k++) acc += sa[k] * fc1_w[k * HH + t];
        sh[t] = 0.5f * acc * (1.0f + erff(acc * 0.70710678118654752f));  // exact gelu
    }
    __syncthreads();
    float p0 = fc2_b[2 * t], p1 = fc2_b[2 * t + 1];
    for (int k = 0; k < HH; k++) {
        const float hv = sh[k];
        p0 += hv * fc2_w[k * (2 * CC) + 2 * t];
        p1 += hv * fc2_w[k * (2 * CC) + 2 * t + 1];
    }
    const float mx = fmaxf(p0, p1);
    const float e0 = expf(p0 - mx), e1 = expf(p1 - mx);
    const float inv = 1.0f / (e0 + e1);
    g0[b * CC + t] = e0 * inv;
    g1[b * CC + t] = e1 * inv;
}

// ---------------------------------------------------------------------------
// Kernel 7: gated mix  out = s*g0 + c*g1  -> bf16 (input to proj GEMM)
// ---------------------------------------------------------------------------
__global__ __launch_bounds__(256)
void mix_kernel(const float* __restrict__ s, const unsigned short* __restrict__ cb,
                const float* __restrict__ g0, const float* __restrict__ g1,
                unsigned short* __restrict__ out) {
    const int i  = blockIdx.x * 256 + threadIdx.x;
    const int ch = i & 255;
    const int b  = i >> 20;
    const float o = s[i] * g0[b * CC + ch] + bfval(cb[i]) * g1[b * CC + ch];
    out[i] = bfbits(o);
}

// ---------------------------------------------------------------------------
extern "C" void kernel_launch(void* const* d_in, const int* in_sizes, int n_in,
                              void* d_out, int out_size, void* d_ws, size_t ws_size,
                              hipStream_t stream) {
    const float* x       = (const float*)d_in[0];
    const int*   mask    = (const int*)d_in[1];
    const float* ln_g    = (const float*)d_in[2];
    const float* ln_b    = (const float*)d_in[3];
    const float* off_w   = (const float*)d_in[4];
    const float* off_b   = (const float*)d_in[5];
    const float* atm_c_w = (const float*)d_in[6];
    const float* atm_n_w = (const float*)d_in[7];
    const float* atm_n_b = (const float*)d_in[8];
    const float* fc1_w   = (const float*)d_in[9];
    const float* fc1_b   = (const float*)d_in[10];
    const float* fc2_w   = (const float*)d_in[11];
    const float* fc2_b   = (const float*)d_in[12];
    const float* proj_w  = (const float*)d_in[13];
    const float* proj_b  = (const float*)d_in[14];
    float* out = (float*)d_out;

    char* ws = (char*)d_ws;
    size_t p = 0;
    auto take = [&](size_t bytes) -> char* {
        p = (p + 255) & ~(size_t)255;
        char* r = ws + p;
        p += bytes;
        return r;
    };
    unsigned short* xn_bf   = (unsigned short*)take((size_t)MM * CC * 2); // reused as mix
    unsigned short* xb_bf   = (unsigned short*)take((size_t)MM * CC * 2);
    unsigned short* samp_bf = (unsigned short*)take((size_t)MM * CC * 2);
    float*          sbuf    = (float*)take((size_t)MM * CC * 4);          // offset, then s
    unsigned short* c_bf    = (unsigned short*)take((size_t)MM * CC * 2);
    unsigned short* wt_off  = (unsigned short*)take(CC * CC * 2);
    unsigned short* wt_n    = (unsigned short*)take(CC * CC * 2);
    unsigned short* wt_c    = (unsigned short*)take(CC * CC * 2);
    unsigned short* wt_p    = (unsigned short*)take(CC * CC * 2);
    float*          abuf    = (float*)take(BB * CC * 4);
    float*          g0      = (float*)take(BB * CC * 4);
    float*          g1      = (float*)take(BB * CC * 4);

    // 1. weights -> bf16 transposed
    prep_weights<<<dim3(CC * CC / 256, 4), 256, 0, stream>>>(
        off_w, atm_n_w, atm_c_w, proj_w, wt_off, wt_n, wt_c, wt_p);
    // 2. LayerNorm + casts
    ln_cast<<<MM / 8, 256, 0, stream>>>(x, ln_g, ln_b, xn_bf, xb_bf);
    // 3. offset = xn @ off_w + off_b  (f32 out, needed for positions)
    gemm_bf16<<<dim3(MM / 256, 2), 256, 0, stream>>>(xn_bf, wt_off, off_b, sbuf, 0);
    // 4. bilinear gather + mask -> bf16
    gather_bilinear<<<MM * CC / 256, 256, 0, stream>>>(x, sbuf, mask, samp_bf);
    // 5. s = sampled @ atm_n_w + atm_n_b  (f32, overwrites offset region)
    gemm_bf16<<<dim3(MM / 256, 2), 256, 0, stream>>>(samp_bf, wt_n, atm_n_b, sbuf, 0);
    // 6. c = x @ atm_c_w  (bf16 out)
    gemm_bf16<<<dim3(MM / 256, 2), 256, 0, stream>>>(xb_bf, wt_c, nullptr, c_bf, 1);
    // 7. masked sum pool (zero + partial atomics)
    zero_f32<<<BB, 256, 0, stream>>>(abuf);
    pool_partial<<<dim3(NN / 256, BB), 256, 0, stream>>>(sbuf, c_bf, mask, abuf);
    // 8. fusion MLP -> gates (divides by mask sum internally)
    fusion_mlp<<<BB, 256, 0, stream>>>(abuf, mask, fc1_w, fc1_b, fc2_w, fc2_b, g0, g1);
    // 9. gated mix -> bf16 (reuse xn region)
    mix_kernel<<<MM * CC / 256, 256, 0, stream>>>(sbuf, c_bf, g0, g1, xn_bf);
    // 10. out = mix @ proj_w + proj_b
    gemm_bf16<<<dim3(MM / 256, 2), 256, 0, stream>>>(xn_bf, wt_p, proj_b, out, 0);
}
